// MultiHeadAttention_1357209665747
// MI455X (gfx1250) — compile-verified, matrix operand order checked by reading, not run on hardware
//
#include <hip/hip_runtime.h>
#include <hip/hip_bf16.h>
#include <math.h>

// ---------------------------------------------------------------------------
// MI455X (gfx1250) multi-head causal attention, bf16 WMMA pipeline.
// Stages: (1) QKV per-head projections  (2) flash-attention (online softmax)
//         (3) output projection + bias.
// Intermediates live in d_ws as bf16 (Q,K,V: [B,H,T,dk], AO: [B,T,D]).
// ---------------------------------------------------------------------------

typedef __attribute__((ext_vector_type(16))) __bf16 v16bf;
typedef __attribute__((ext_vector_type(8)))  float  v8f;

#define MHA_B  2
#define MHA_T  2048
#define MHA_D  1024
#define MHA_H  16
#define MHA_DK 64

// K-dim index for 16-bit A/B WMMA fragments (ISA 7.12.2):
// element e of lane l holds K = (e<8 ? e : e+8) + 8*(lane>=16)
__device__ __forceinline__ int kmap(int e, int lane) {
  return (e < 8 ? e : e + 8) + ((lane >> 4) << 3);
}

__device__ __forceinline__ v8f wmma_bf16(v16bf a, v16bf b, v8f c) {
  return __builtin_amdgcn_wmma_f32_16x16x32_bf16(
      /*neg_a=*/false, a, /*neg_b=*/false, b,
      /*c_mod=*/(short)0, c, /*reuse_a=*/false, /*reuse_b=*/false);
}

// ---------------------------------------------------------------------------
// Kernel 1: per-head QKV projection.  y = x_h @ W^T  (W is [dk][dk] row-major)
// grid: (T/16, B*H), block: 128 (4 waves; wave w owns N-tile w of dk=64)
// ---------------------------------------------------------------------------
__global__ void qkv_proj_kernel(const float* __restrict__ x,
                                const float* __restrict__ wq,
                                const float* __restrict__ wk,
                                const float* __restrict__ wv,
                                __bf16* __restrict__ q,
                                __bf16* __restrict__ k,
                                __bf16* __restrict__ v) {
  const int tid  = threadIdx.x;
  const int wave = tid >> 5;
  const int lane = tid & 31;
  const int lm   = lane & 15;
  const int hi   = lane >> 4;

  const int t0 = blockIdx.x * 16;
  const int bh = blockIdx.y;            // b*H + h
  const int b  = bh / MHA_H;
  const int h  = bh % MHA_H;
  const int n0 = wave * 16;

  // A fragments: rows of x_h (16 x 64, two K-steps of 32)
  v16bf a0, a1;
  {
    const size_t base = ((size_t)b * MHA_T + (t0 + lm)) * MHA_D + h * MHA_DK;
#pragma unroll
    for (int e = 0; e < 16; ++e) {
      const int kk = kmap(e, lane);
      a0[e] = (__bf16)x[base + kk];
      a1[e] = (__bf16)x[base + 32 + kk];
    }
  }

  // B fragments from each weight: B[k][n] = W[n][k] -> row-major row gather
  const size_t wbase = ((size_t)h * MHA_DK + (n0 + lm)) * MHA_DK;
  v16bf bq0, bq1, bk0, bk1, bv0, bv1;
#pragma unroll
  for (int e = 0; e < 16; ++e) {
    const int kk = kmap(e, lane);
    bq0[e] = (__bf16)wq[wbase + kk];       bq1[e] = (__bf16)wq[wbase + 32 + kk];
    bk0[e] = (__bf16)wk[wbase + kk];       bk1[e] = (__bf16)wk[wbase + 32 + kk];
    bv0[e] = (__bf16)wv[wbase + kk];       bv1[e] = (__bf16)wv[wbase + 32 + kk];
  }

  v8f cq = {}, ck = {}, cv = {};
  cq = wmma_bf16(a0, bq0, cq);  cq = wmma_bf16(a1, bq1, cq);
  ck = wmma_bf16(a0, bk0, ck);  ck = wmma_bf16(a1, bk1, ck);
  cv = wmma_bf16(a0, bv0, cv);  cv = wmma_bf16(a1, bv1, cv);

  // C/D layout: element r -> row M = r + 8*hi, col N = lm
  const size_t obase = ((size_t)bh * MHA_T + t0) * MHA_DK;
#pragma unroll
  for (int r = 0; r < 8; ++r) {
    const int M = r + 8 * hi;
    const size_t idx = obase + (size_t)M * MHA_DK + n0 + lm;
    q[idx] = (__bf16)cq[r];
    k[idx] = (__bf16)ck[r];
    v[idx] = (__bf16)cv[r];
  }
}

// ---------------------------------------------------------------------------
// Kernel 2: causal flash attention, one wave per (b,h, 16-row Q tile).
// K/V consumed in 32-column blocks; online softmax; P relaid out via LDS.
// grid: (T/16, B*H), block: 32
// ---------------------------------------------------------------------------
__global__ void flash_attn_kernel(const __bf16* __restrict__ q,
                                  const __bf16* __restrict__ k,
                                  const __bf16* __restrict__ v,
                                  __bf16* __restrict__ ao) {
  __shared__ __bf16 lds_p[16 * 32];      // P tile (16 rows x 32 cols)

  const int lane = threadIdx.x & 31;
  const int lm   = lane & 15;
  const int hi   = lane >> 4;
  const int t0   = blockIdx.x * 16;
  const int bh   = blockIdx.y;
  const int b    = bh / MHA_H;
  const int h    = bh % MHA_H;

  // Q A-fragments (16 x 64 -> two K-steps)
  v16bf qa0, qa1;
  {
    const size_t base = ((size_t)bh * MHA_T + (t0 + lm)) * MHA_DK;
#pragma unroll
    for (int e = 0; e < 16; ++e) {
      const int kk = kmap(e, lane);
      qa0[e] = q[base + kk];
      qa1[e] = q[base + 32 + kk];
    }
  }

  v8f O0 = {}, O1 = {}, O2 = {}, O3 = {};
  float m_r[8], l_r[8];
#pragma unroll
  for (int r = 0; r < 8; ++r) { m_r[r] = -3.0e38f; l_r[r] = 0.0f; }

  const float scale = 0.125f;            // 1/sqrt(64)
  const int jmax = (t0 + 15) / 32;       // causal: last 32-col block touching tile

  for (int j = 0; j <= jmax; ++j) {
    const int c0 = 32 * j;

    // K B-fragments: B[kk][n] = k_row(c0 + tile*16 + n)[kk]
    v16bf kb00, kb01, kb10, kb11;
    {
      const size_t r0 = ((size_t)bh * MHA_T + (c0 + lm)) * MHA_DK;
      const size_t r1 = ((size_t)bh * MHA_T + (c0 + 16 + lm)) * MHA_DK;
#pragma unroll
      for (int e = 0; e < 16; ++e) {
        const int kk = kmap(e, lane);
        kb00[e] = k[r0 + kk];   kb01[e] = k[r0 + 32 + kk];
        kb10[e] = k[r1 + kk];   kb11[e] = k[r1 + 32 + kk];
      }
    }

    v8f S0 = {}, S1 = {};
    S0 = wmma_bf16(qa0, kb00, S0);  S0 = wmma_bf16(qa1, kb01, S0);
    S1 = wmma_bf16(qa0, kb10, S1);  S1 = wmma_bf16(qa1, kb11, S1);

    // mask + online softmax (row M = r + 8*hi lives across 16-lane half)
#pragma unroll
    for (int r = 0; r < 8; ++r) {
      const int row  = t0 + r + 8 * hi;
      const int col0 = c0 + lm;
      const int col1 = c0 + 16 + lm;
      float s0 = S0[r] * scale;  if (col0 > row) s0 = -3.0e38f;
      float s1 = S1[r] * scale;  if (col1 > row) s1 = -3.0e38f;

      float mx = fmaxf(s0, s1);
#pragma unroll
      for (int off = 1; off < 16; off <<= 1)
        mx = fmaxf(mx, __shfl_xor(mx, off, 32));

      const float mnew  = fmaxf(m_r[r], mx);
      const float alpha = __expf(m_r[r] - mnew);
      const float p0 = __expf(s0 - mnew);
      const float p1 = __expf(s1 - mnew);

      float rs = p0 + p1;
#pragma unroll
      for (int off = 1; off < 16; off <<= 1)
        rs += __shfl_xor(rs, off, 32);

      l_r[r] = l_r[r] * alpha + rs;
      m_r[r] = mnew;
      O0[r] *= alpha;  O1[r] *= alpha;  O2[r] *= alpha;  O3[r] *= alpha;

      const int M = r + 8 * hi;
      lds_p[M * 32 + lm]      = (__bf16)p0;
      lds_p[M * 32 + 16 + lm] = (__bf16)p1;
    }
    __syncthreads();

    // Re-stripe P (C layout) into A layout via LDS
    v16bf pa;
#pragma unroll
    for (int e = 0; e < 16; ++e)
      pa[e] = lds_p[lm * 32 + kmap(e, lane)];

    // V B-fragments: B[kk][n] = v_row(c0+kk)[nt*16 + n]
    v16bf vb;
    const size_t vbase = (size_t)bh * MHA_T * MHA_DK;
#pragma unroll
    for (int e = 0; e < 16; ++e)
      vb[e] = v[vbase + (size_t)(c0 + kmap(e, lane)) * MHA_DK + lm];
    O0 = wmma_bf16(pa, vb, O0);
#pragma unroll
    for (int e = 0; e < 16; ++e)
      vb[e] = v[vbase + (size_t)(c0 + kmap(e, lane)) * MHA_DK + 16 + lm];
    O1 = wmma_bf16(pa, vb, O1);
#pragma unroll
    for (int e = 0; e < 16; ++e)
      vb[e] = v[vbase + (size_t)(c0 + kmap(e, lane)) * MHA_DK + 32 + lm];
    O2 = wmma_bf16(pa, vb, O2);
#pragma unroll
    for (int e = 0; e < 16; ++e)
      vb[e] = v[vbase + (size_t)(c0 + kmap(e, lane)) * MHA_DK + 48 + lm];
    O3 = wmma_bf16(pa, vb, O3);

    __syncthreads();
  }

  // normalize + store as [B, T, H*dk] (= concat heads) bf16
#pragma unroll
  for (int r = 0; r < 8; ++r) {
    const int M = r + 8 * hi;
    const float inv = 1.0f / l_r[r];
    const size_t base = ((size_t)b * MHA_T + (t0 + M)) * MHA_D + h * MHA_DK + lm;
    ao[base]      = (__bf16)(O0[r] * inv);
    ao[base + 16] = (__bf16)(O1[r] * inv);
    ao[base + 32] = (__bf16)(O2[r] * inv);
    ao[base + 48] = (__bf16)(O3[r] * inv);
  }
}

// ---------------------------------------------------------------------------
// Kernel 3: output projection  out = AO @ Wp^T + bp   (fp32 out)
// grid: (B*T/16, D/64), block: 128 (4 waves; wave w owns one 16-wide N-tile)
// A tile staged in LDS and reused by all 4 waves.
// ---------------------------------------------------------------------------
__global__ void out_proj_kernel(const __bf16* __restrict__ ao,
                                const float* __restrict__ wp,
                                const float* __restrict__ bp,
                                float* __restrict__ out) {
  __shared__ __bf16 lds_a[16 * 32];

  const int tid  = threadIdx.x;
  const int wave = tid >> 5;
  const int lane = tid & 31;
  const int lm   = lane & 15;
  const int hi   = lane >> 4;

  const int m0 = blockIdx.x * 16;
  const int n0 = blockIdx.y * 64 + wave * 16;

  v8f acc = {};
  for (int s = 0; s < MHA_D / 32; ++s) {
    const int kk0 = s * 32;

    // cooperative stage of A tile (16 x 32 bf16, 512 elems, 4 per thread)
#pragma unroll
    for (int u = 0; u < 4; ++u) {
      const int idx = tid * 4 + u;
      lds_a[idx] = ao[(size_t)(m0 + (idx >> 5)) * MHA_D + kk0 + (idx & 31)];
    }
    if (s + 1 < MHA_D / 32) {  // prefetch next A tile (global_prefetch_b8)
      const int idx = tid * 4;
      __builtin_prefetch(&ao[(size_t)(m0 + (idx >> 5)) * MHA_D + kk0 + 32 + (idx & 31)], 0, 1);
    }
    __syncthreads();

    v16bf a, bfr;
#pragma unroll
    for (int e = 0; e < 16; ++e) {
      const int kk = kmap(e, lane);
      a[e]   = lds_a[lm * 32 + kk];
      bfr[e] = (__bf16)wp[(size_t)(n0 + lm) * MHA_D + kk0 + kk];
    }
    acc = wmma_bf16(a, bfr, acc);
    __syncthreads();
  }

  const float bias = bp[n0 + lm];
#pragma unroll
  for (int r = 0; r < 8; ++r) {
    const int M = r + 8 * hi;
    out[(size_t)(m0 + M) * MHA_D + n0 + lm] = acc[r] + bias;
  }
}

// ---------------------------------------------------------------------------
extern "C" void kernel_launch(void* const* d_in, const int* in_sizes, int n_in,
                              void* d_out, int out_size, void* d_ws, size_t ws_size,
                              hipStream_t stream) {
  const float* x  = (const float*)d_in[0];
  const float* wq = (const float*)d_in[1];
  const float* wk = (const float*)d_in[2];
  const float* wv = (const float*)d_in[3];
  const float* wp = (const float*)d_in[4];
  const float* bp = (const float*)d_in[5];
  float* out = (float*)d_out;

  // workspace: Q, K, V [B,H,T,dk] bf16 + AO [B,T,D] bf16  (4 x 8 MB = 32 MB)
  const size_t NELEM = (size_t)MHA_B * MHA_H * MHA_T * MHA_DK;  // 4,194,304
  __bf16* q  = (__bf16*)d_ws;
  __bf16* k  = q + NELEM;
  __bf16* v  = k + NELEM;
  __bf16* ao = v + NELEM;

  qkv_proj_kernel<<<dim3(MHA_T / 16, MHA_B * MHA_H), 128, 0, stream>>>(
      x, wq, wk, wv, q, k, v);
  flash_attn_kernel<<<dim3(MHA_T / 16, MHA_B * MHA_H), 32, 0, stream>>>(
      q, k, v, ao);
  out_proj_kernel<<<dim3(MHA_B * MHA_T / 16, MHA_D / 64), 128, 0, stream>>>(
      ao, wp, bp, out);
}
